// NeuroMCPHead_69011534512402
// MI455X (gfx1250) — matrix operand order repaired and checked by reading, not compiled, not run on hardware
//
#include <hip/hip_runtime.h>
#include <hip/hip_bf16.h>

typedef __attribute__((ext_vector_type(16))) __bf16 v16bf;
typedef __attribute__((ext_vector_type(4)))  __bf16 v4bf;
typedef __attribute__((ext_vector_type(8)))  float  v8f;

#define HIDDEN 1024
#define MROWS  32768
#define KALIVE 9

// B tile staging in LDS: 128 cols x 32 K bf16, 80-byte column stride (64B data
// + 16B pad -> start banks walk 20 apart, no 4-way conflicts), hi + lo planes,
// double buffered.
#define COLSTRIDE 80u
#define PLANE     (128u * COLSTRIDE)   // 10240 B
#define BUFBYTES  (2u * PLANE)         // hi+lo: 20480 B
// total LDS: 2 * BUFBYTES = 40960 B (of 320 KB per WGP)

// ---------------------------------------------------------------------------
// Kernel 1: masked weight -> bf16 hi/lo split planes (for bf16x3 GEMM)
// ---------------------------------------------------------------------------
__global__ __launch_bounds__(256) void k_wprep(const float* __restrict__ w,
                                               const float* __restrict__ m,
                                               __bf16* __restrict__ Wh,
                                               __bf16* __restrict__ Wl) {
    size_t i = ((size_t)blockIdx.x * 256 + threadIdx.x) * 4;
    float4 a = *(const float4*)(w + i);
    float4 b = *(const float4*)(m + i);
    float x[4] = {a.x * b.x, a.y * b.y, a.z * b.z, a.w * b.w};
    v4bf hi, lo;
#pragma unroll
    for (int j = 0; j < 4; ++j) {
        __bf16 hb = (__bf16)x[j];
        hi[j] = hb;
        lo[j] = (__bf16)(x[j] - (float)hb);
    }
    *(v4bf*)(Wh + i) = hi;
    *(v4bf*)(Wl + i) = lo;
}

// ---------------------------------------------------------------------------
// Kernel 2: L2-normalize the 9 live prototype rows (one wave32 per row)
// ---------------------------------------------------------------------------
__global__ __launch_bounds__(288) void k_pprep(const float* __restrict__ proto,
                                               float* __restrict__ protoN) {
    const int lane = threadIdx.x & 31;
    const int wave = threadIdx.x >> 5;   // 9 waves, one per prototype row
    const float* pr = proto + (size_t)wave * HIDDEN;
    float4 v[8];
    float ss = 0.f;
#pragma unroll
    for (int c = 0; c < 8; ++c) {
        v[c] = *(const float4*)(pr + c * 128 + lane * 4);
        ss += v[c].x * v[c].x + v[c].y * v[c].y + v[c].z * v[c].z + v[c].w * v[c].w;
    }
#pragma unroll
    for (int m = 16; m; m >>= 1) ss += __shfl_xor(ss, m, 32);
    const float inv = 1.f / fmaxf(sqrtf(ss), 1e-12f);
    float* out = protoN + (size_t)wave * HIDDEN;
#pragma unroll
    for (int c = 0; c < 8; ++c) {
        float4 o = {v[c].x * inv, v[c].y * inv, v[c].z * inv, v[c].w * inv};
        *(float4*)(out + c * 128 + lane * 4) = o;
    }
}

// ---------------------------------------------------------------------------
// Async copy: one 16B chunk per lane into LDS (GLOBAL_LOAD_ASYNC_TO_LDS_B128,
// ASYNCcnt-tracked, no VGPR landing zone).
// ---------------------------------------------------------------------------
__device__ inline void async_b128(unsigned lds, uint64_t gaddr) {
    asm volatile("global_load_async_to_lds_b128 %0, %1, off"
                 :: "v"(lds), "v"(gaddr) : "memory");
}

// Issue one K-step B tile (128 cols x 32 K, hi+lo planes) = 4 async b128 ops.
// 256 lanes x 16B = 4 KB per op; lane t -> col t>>2 (+64 for second op), K
// chunk (t&3)*8 elements.
__device__ inline void issue_b(const __bf16* __restrict__ Wh,
                               const __bf16* __restrict__ Wl,
                               unsigned bufOff, int blockN, int k0, int tid) {
    const int colLocal = tid >> 2;            // 0..63
    const int chunk    = tid & 3;             // 0..3
    const unsigned ldsA = bufOff + (unsigned)colLocal * COLSTRIDE + (unsigned)chunk * 16u;
    const size_t   gEl  = (size_t)(blockN + colLocal) * HIDDEN + k0 + chunk * 8;
    async_b128(ldsA,                    (uint64_t)(uintptr_t)(Wh + gEl));
    async_b128(ldsA + 64u * COLSTRIDE,  (uint64_t)(uintptr_t)(Wh + gEl + (size_t)64 * HIDDEN));
    async_b128(ldsA + PLANE,                   (uint64_t)(uintptr_t)(Wl + gEl));
    async_b128(ldsA + PLANE + 64u * COLSTRIDE, (uint64_t)(uintptr_t)(Wl + gEl + (size_t)64 * HIDDEN));
}

// ---------------------------------------------------------------------------
// Kernel 3: z_raw = h @ Wm^T via v_wmma_f32_16x16x32_bf16 (bf16x3 split).
// Block tile 128x128 (8 waves = 4M x 2N), wave tile 32x64 (2M x 4N tiles),
// K-step 32, B double-buffered in LDS via async loads.
// ---------------------------------------------------------------------------
__device__ inline void cvt8(const float4& a, const float4& b,
                            v16bf& hi, v16bf& lo, int base) {
    float xs[8] = {a.x, a.y, a.z, a.w, b.x, b.y, b.z, b.w};
#pragma unroll
    for (int i = 0; i < 8; ++i) {
        float x = xs[i];
        __bf16 hb = (__bf16)x;
        hi[base + i] = hb;
        lo[base + i] = (__bf16)(x - (float)hb);
    }
}

__global__ __launch_bounds__(256) void k_gemm(const float* __restrict__ h,
                                              const __bf16* __restrict__ Wh,
                                              const __bf16* __restrict__ Wl,
                                              float* __restrict__ zraw) {
    __shared__ __align__(16) unsigned char smem[2 * BUFBYTES];

    const int tid   = threadIdx.x;
    const int lane  = tid & 31;
    const int wave  = tid >> 5;
    const int waveM = wave & 3;          // 4 waves along M
    const int waveN = wave >> 2;         // 2 waves along N
    const int blockM = blockIdx.x * 128;
    const int blockN = blockIdx.y * 128;
    const int rowA = blockM + waveM * 32 + (lane & 15);
    const int kHalf = lane >> 4;
    const int kA    = kHalf * 16;        // A: second lane-half starts at K+16
    // per-lane byte offset of this wave's B columns inside a plane
    const unsigned bOffLane =
        (unsigned)(waveN * 64 + (lane & 15)) * COLSTRIDE + (unsigned)kHalf * 32u;
    const unsigned smemBase = (unsigned)(size_t)&smem[0];

    v8f zero = {};
    v8f acc[2][4] = {{zero, zero, zero, zero}, {zero, zero, zero, zero}};

    const int steps = HIDDEN / 32;       // 32
    issue_b(Wh, Wl, smemBase, blockN, 0, tid);

    for (int i = 0; i < steps; ++i) {
        const int k0 = i * 32;
        if (i + 1 < steps) {
            issue_b(Wh, Wl, smemBase + (unsigned)((i + 1) & 1) * BUFBYTES,
                    blockN, k0 + 32, tid);
            // async loads complete in order: <=4 outstanding means the current
            // buffer's 4 issues are done, next buffer's 4 still in flight
            asm volatile("s_wait_asynccnt 4" ::: "memory");
        } else {
            asm volatile("s_wait_asynccnt 0" ::: "memory");
        }
        __syncthreads();

        const unsigned char* buf = smem + (unsigned)(i & 1) * BUFBYTES;

        // A fragments: direct global loads + in-register bf16 hi/lo split
        v16bf ah[2], al[2];
#pragma unroll
        for (int mt = 0; mt < 2; ++mt) {
            const float* pa = h + (size_t)(rowA + mt * 16) * HIDDEN + k0 + kA;
            float4 a0 = *(const float4*)(pa);
            float4 a1 = *(const float4*)(pa + 4);
            float4 a2 = *(const float4*)(pa + 8);
            float4 a3 = *(const float4*)(pa + 12);
            __builtin_prefetch(pa + 32, 0, 0);   // global_prefetch_b8, next K tile
            cvt8(a0, a1, ah[mt], al[mt], 0);
            cvt8(a2, a3, ah[mt], al[mt], 8);
        }

#pragma unroll
        for (int nt = 0; nt < 4; ++nt) {
            v16bf bh = *(const v16bf*)(buf + bOffLane + (unsigned)nt * (16u * COLSTRIDE));
            v16bf bl = *(const v16bf*)(buf + PLANE + bOffLane + (unsigned)nt * (16u * COLSTRIDE));
#pragma unroll
            for (int mt = 0; mt < 2; ++mt) {
                acc[mt][nt] = __builtin_amdgcn_wmma_f32_16x16x32_bf16(
                    false, ah[mt], false, bh, (short)0, acc[mt][nt], false, false);
                acc[mt][nt] = __builtin_amdgcn_wmma_f32_16x16x32_bf16(
                    false, ah[mt], false, bl, (short)0, acc[mt][nt], false, false);
                acc[mt][nt] = __builtin_amdgcn_wmma_f32_16x16x32_bf16(
                    false, al[mt], false, bh, (short)0, acc[mt][nt], false, false);
            }
        }
        __syncthreads();   // protect buffer (i&1) before it is overwritten
    }

    // C/D layout: VGPR i -> M = i (lanes 0..15) or i+8 (lanes 16..31), N = lane&15
    const int rbase = blockM + waveM * 32 + ((lane < 16) ? 0 : 8);
    const int cbase = blockN + waveN * 64 + (lane & 15);
#pragma unroll
    for (int mt = 0; mt < 2; ++mt)
#pragma unroll
        for (int nt = 0; nt < 4; ++nt)
#pragma unroll
            for (int i = 0; i < 8; ++i)
                zraw[(size_t)(rbase + mt * 16 + i) * HIDDEN + (cbase + nt * 16)] =
                    acc[mt][nt][i];
}

// ---------------------------------------------------------------------------
// Kernel 4: in-place LayerNorm + cosine logits. One wave32 per row.
// ---------------------------------------------------------------------------
__global__ __launch_bounds__(256) void k_ln(float* __restrict__ z,
                                            float* __restrict__ logits,
                                            const float* __restrict__ gamma,
                                            const float* __restrict__ beta,
                                            const float* __restrict__ protoN) {
    const int lane = threadIdx.x & 31;
    const int wave = threadIdx.x >> 5;
    const int row  = blockIdx.x * 8 + wave;
    float* zr = z + (size_t)row * HIDDEN;

    float4 v[8];
    float sum = 0.f, sumsq = 0.f;
#pragma unroll
    for (int c = 0; c < 8; ++c) {
        v[c] = *(const float4*)(zr + c * 128 + lane * 4);
        sum   += v[c].x + v[c].y + v[c].z + v[c].w;
        sumsq += v[c].x * v[c].x + v[c].y * v[c].y + v[c].z * v[c].z + v[c].w * v[c].w;
    }
#pragma unroll
    for (int m = 16; m; m >>= 1) {
        sum   += __shfl_xor(sum, m, 32);
        sumsq += __shfl_xor(sumsq, m, 32);
    }
    const float mean = sum * (1.f / 1024.f);
    const float var  = sumsq * (1.f / 1024.f) - mean * mean;
    const float rstd = rsqrtf(var + 1e-5f);

    float nrm2 = 0.f;
#pragma unroll
    for (int c = 0; c < 8; ++c) {
        float4 g  = *(const float4*)(gamma + c * 128 + lane * 4);
        float4 bt = *(const float4*)(beta  + c * 128 + lane * 4);
        v[c].x = (v[c].x - mean) * rstd * g.x + bt.x;
        v[c].y = (v[c].y - mean) * rstd * g.y + bt.y;
        v[c].z = (v[c].z - mean) * rstd * g.z + bt.z;
        v[c].w = (v[c].w - mean) * rstd * g.w + bt.w;
        nrm2 += v[c].x * v[c].x + v[c].y * v[c].y + v[c].z * v[c].z + v[c].w * v[c].w;
        *(float4*)(zr + c * 128 + lane * 4) = v[c];
    }
#pragma unroll
    for (int m = 16; m; m >>= 1) nrm2 += __shfl_xor(nrm2, m, 32);
    const float inv = 1.f / (fmaxf(sqrtf(nrm2), 1e-12f) * 0.07f);

    float dots[KALIVE];
#pragma unroll
    for (int k = 0; k < KALIVE; ++k) {
        const float* p = protoN + (size_t)k * HIDDEN;
        float d = 0.f;
#pragma unroll
        for (int c = 0; c < 8; ++c) {
            float4 pk = *(const float4*)(p + c * 128 + lane * 4);
            d += v[c].x * pk.x + v[c].y * pk.y + v[c].z * pk.z + v[c].w * pk.w;
        }
#pragma unroll
        for (int m = 16; m; m >>= 1) d += __shfl_xor(d, m, 32);
        dots[k] = d * inv;
    }
    if (lane == 0) {
        float* lo = logits + (size_t)row * KALIVE;
#pragma unroll
        for (int k = 0; k < KALIVE; ++k) lo[k] = dots[k];
    }
}

// ---------------------------------------------------------------------------
extern "C" void kernel_launch(void* const* d_in, const int* in_sizes, int n_in,
                              void* d_out, int out_size, void* d_ws, size_t ws_size,
                              hipStream_t stream) {
    const float* h      = (const float*)d_in[0];
    const float* weight = (const float*)d_in[1];
    const float* mask   = (const float*)d_in[2];
    const float* gamma  = (const float*)d_in[3];
    const float* beta   = (const float*)d_in[4];
    const float* proto  = (const float*)d_in[5];

    // Workspace: Wh (2MB) | Wl (2MB) | protoN (36KB)
    __bf16* Wh = (__bf16*)d_ws;
    __bf16* Wl = Wh + (size_t)HIDDEN * HIDDEN;
    float*  protoN = (float*)(Wl + (size_t)HIDDEN * HIDDEN);

    // d_out layout: logits (32768 x 9) then z (32768 x 1024)
    float* logits = (float*)d_out;
    float* zout   = logits + (size_t)MROWS * KALIVE;

    k_wprep<<<dim3(HIDDEN * HIDDEN / (256 * 4)), dim3(256), 0, stream>>>(weight, mask, Wh, Wl);
    k_pprep<<<dim3(1), dim3(288), 0, stream>>>(proto, protoN);
    k_gemm<<<dim3(MROWS / 128, HIDDEN / 128), dim3(256), 0, stream>>>(h, Wh, Wl, zout);
    k_ln<<<dim3(MROWS / 8), dim3(256), 0, stream>>>(zout, logits, gamma, beta, protoN);
}
